// GraphIsomorphismNetwork_81028853006932
// MI455X (gfx1250) — matrix operand order repaired and checked by reading, not compiled, not run on hardware
//
#include <hip/hip_runtime.h>
#include <hip/hip_bf16.h>

// ---------------- problem constants (match reference) ----------------
#define NN 50000
#define EE 800000
#define DD 128
#define GG 256
#define CC 16
#define LL 5
#define ND (NN * DD)   // 6,400,000

typedef __attribute__((ext_vector_type(16))) __bf16 v16bf;
typedef __attribute__((ext_vector_type(8)))  float  v8f;
typedef __attribute__((ext_vector_type(8)))  __bf16 bf16x8;
typedef __attribute__((ext_vector_type(16))) __bf16 bf16x16;
typedef __attribute__((ext_vector_type(4)))  __bf16 bf16x4;

// ---------------- elementwise / utility kernels ----------------

__global__ void k_zero(float* p, int n) {
    int i = blockIdx.x * blockDim.x + threadIdx.x;
    if (i < n) p[i] = 0.0f;
}

// pooled = (1 + eps[l]) * h
__global__ void k_init_pooled(const float* __restrict__ h,
                              const float* __restrict__ eps, int l,
                              float* __restrict__ pooled) {
    int i = blockIdx.x * blockDim.x + threadIdx.x;   // grid sized exactly ND
    float e = 1.0f + eps[l];
    pooled[i] = e * h[i];
}

// scatter-add over edges: pooled[dst] += h[src]  (E*D threads)
__global__ void k_edge_scatter(const float* __restrict__ h,
                               const int* __restrict__ src,
                               const int* __restrict__ dst,
                               float* __restrict__ pooled) {
    long long idx = (long long)blockIdx.x * blockDim.x + threadIdx.x; // E*D
    int e = (int)(idx >> 7);
    int d = (int)(idx & (DD - 1));
    int s  = src[e];
    int t  = dst[e];
    float v = h[(long long)s * DD + d];
    unsafeAtomicAdd(&pooled[(long long)t * DD + d], v);   // global_atomic_add_f32
}

// fp32 -> bf16, 4 elements per thread (ND/4 threads)
__global__ void k_cvt_bf16x4(const float* __restrict__ in, __bf16* __restrict__ out) {
    int i = blockIdx.x * blockDim.x + threadIdx.x;
    const float4 v = ((const float4*)in)[i];
    bf16x4 o;
    o.x = (__bf16)v.x; o.y = (__bf16)v.y; o.z = (__bf16)v.z; o.w = (__bf16)v.w;
    ((bf16x4*)out)[i] = o;
}

// W (DxD row-major, [k][n]) -> bf16 transposed Wt[n*D + k]  (D*D threads)
__global__ void k_cvt_w_transpose(const float* __restrict__ W, __bf16* __restrict__ Wt) {
    int i = blockIdx.x * blockDim.x + threadIdx.x;   // D*D
    int k = i >> 7;
    int n = i & (DD - 1);
    Wt[n * DD + k] = (__bf16)W[k * DD + n];
}

// ---------------- bf16 WMMA GEMM: C = A(Nx128) * B(128x128) + bias ----------------
// A: bf16 row-major.  Bt: bf16 with columns of B contiguous (Bt[n*128+k]).
// One wave -> one 16x16 tile, K-loop of 4 x v_wmma_f32_16x16x32_bf16.
__global__ __launch_bounds__(128) void k_gemm_bf16(const __bf16* __restrict__ A,
                                                   const __bf16* __restrict__ Bt,
                                                   const float* __restrict__ bias,
                                                   float* __restrict__ Cmat,
                                                   int Mtiles) {
    int wave = threadIdx.x >> 5;
    int lane = threadIdx.x & 31;
    int tile = blockIdx.x * 4 + wave;        // 4 waves/block share the same M-tile
    int mt = tile >> 3;                      // 8 n-tiles per m-tile (D=128)
    int nt = tile & 7;
    if (mt >= Mtiles) return;                // wave-uniform; EXEC all-1s inside

    int r  = lane & 15;                      // row (A) / column (B,C) within tile
    int kh = lane >> 4;                      // K half selector

    const __bf16* Arow = A  + (size_t)(mt * 16 + r) * DD;
    const __bf16* Bcol = Bt + (size_t)(nt * 16 + r) * DD;

    v8f acc = {};
#pragma unroll
    for (int ks = 0; ks < 4; ++ks) {
        int kb = ks * 32;
        // A 16x32 fragment: lane half kh holds K = kb+kh*8+{0..7} and kb+16+kh*8+{0..7}
        bf16x8 a0 = *(const bf16x8*)(Arow + kb + kh * 8);
        bf16x8 a1 = *(const bf16x8*)(Arow + kb + 16 + kh * 8);
        // B 32x16 fragment: lane half kh holds K = kb+kh*8+{0..7} and kb+16+kh*8+{0..7} at fixed column
        bf16x8 b0 = *(const bf16x8*)(Bcol + kb + kh * 8);
        bf16x8 b1 = *(const bf16x8*)(Bcol + kb + 16 + kh * 8);
        v16bf a, b;
#pragma unroll
        for (int j = 0; j < 8; ++j) { a[j] = a0[j]; a[8 + j] = a1[j]; }
#pragma unroll
        for (int j = 0; j < 8; ++j) { b[j] = b0[j]; b[8 + j] = b1[j]; }
        acc = __builtin_amdgcn_wmma_f32_16x16x32_bf16(
            /*neg_a=*/false, a, /*neg_b=*/false, b,
            /*c_mod=*/(short)0, acc, /*reuse_a=*/false, /*reuse_b=*/false);
    }

    int n  = nt * 16 + r;                    // output column for this lane
    float bc = bias[n];
#pragma unroll
    for (int rr = 0; rr < 8; ++rr) {         // VGPR rr -> M = kh*8 + rr
        int m = mt * 16 + kh * 8 + rr;
        Cmat[(size_t)m * DD + n] = acc[rr] + bc;
    }
}

// ---------------- BatchNorm (training-mode batch stats over N) ----------------

// partial column sums / sums-of-squares, block = 2 rows x 128 cols
__global__ void k_colstats(const float* __restrict__ z, float* __restrict__ stats, int Nrows) {
    int col   = threadIdx.x & (DD - 1);
    int rlane = threadIdx.x >> 7;            // 0..1
    float s = 0.0f, ss = 0.0f;
    for (int r = blockIdx.x * 2 + rlane; r < Nrows; r += gridDim.x * 2) {
        float v = z[(size_t)r * DD + col];
        s += v; ss += v * v;
    }
    unsafeAtomicAdd(&stats[col], s);
    unsafeAtomicAdd(&stats[DD + col], ss);
}

// stats -> (scale, shift): scale = g*rsqrt(var+eps), shift = beta - mean*scale
__global__ void k_bn_finalize(float* __restrict__ stats,
                              const float* __restrict__ g,
                              const float* __restrict__ beta, float invN) {
    int col = threadIdx.x;                   // 128 threads
    float mean = stats[col] * invN;
    float var  = stats[DD + col] * invN - mean * mean;
    float sc   = g[col] * rsqrtf(var + 1e-5f);
    stats[col]      = sc;
    stats[DD + col] = beta[col] - mean * sc;
}

// y = relu(z*scale+shift) -> bf16 (feeds next GEMM)
__global__ void k_bn_relu_bf16(const float* __restrict__ z,
                               const float* __restrict__ stats,
                               __bf16* __restrict__ out) {
    int i = blockIdx.x * blockDim.x + threadIdx.x;   // ND
    int col = i & (DD - 1);
    float v = z[i] * stats[col] + stats[DD + col];
    out[i] = (__bf16)fmaxf(v, 0.0f);
}

// y = relu(z*scale+shift) -> fp32 (new h)
__global__ void k_bn_relu_f32(const float* __restrict__ z,
                              const float* __restrict__ stats,
                              float* __restrict__ out) {
    int i = blockIdx.x * blockDim.x + threadIdx.x;   // ND
    int col = i & (DD - 1);
    float v = z[i] * stats[col] + stats[DD + col];
    out[i] = fmaxf(v, 0.0f);
}

// ---------------- graph readout ----------------

// gpool[graph_ids[node]] += h[node]  (N*D threads)
__global__ void k_graph_pool(const float* __restrict__ h,
                             const int* __restrict__ gids,
                             float* __restrict__ gpool) {
    int idx = blockIdx.x * blockDim.x + threadIdx.x; // ND
    int node = idx >> 7;
    int d    = idx & (DD - 1);
    int g = gids[node];
    unsafeAtomicAdd(&gpool[g * DD + d], h[idx]);
}

// out[g][c] += dot(gpool[g], Wp[:,c]) + bp[c]   (G*C = 4096 threads)
__global__ void k_proj(const float* __restrict__ gpool,
                       const float* __restrict__ Wp,   // DxC row-major
                       const float* __restrict__ bp,   // C
                       float* __restrict__ out) {
    int idx = blockIdx.x * blockDim.x + threadIdx.x;
    int g = idx >> 4;
    int c = idx & (CC - 1);
    float s = bp[c];
    const float* gp = gpool + g * DD;
#pragma unroll 8
    for (int k = 0; k < DD; ++k) s += gp[k] * Wp[k * CC + c];
    out[idx] += s;
}

// ---------------- host orchestration ----------------

extern "C" void kernel_launch(void* const* d_in, const int* in_sizes, int n_in,
                              void* d_out, int out_size, void* d_ws, size_t ws_size,
                              hipStream_t stream) {
    const float* x    = (const float*)d_in[0];
    const float* eps  = (const float*)d_in[1];
    const float* W1   = (const float*)d_in[2];
    const float* b1   = (const float*)d_in[3];
    const float* g1   = (const float*)d_in[4];
    const float* be1  = (const float*)d_in[5];
    const float* W2   = (const float*)d_in[6];
    const float* b2   = (const float*)d_in[7];
    const float* g2   = (const float*)d_in[8];
    const float* be2  = (const float*)d_in[9];
    const float* Wp   = (const float*)d_in[10];
    const float* bp   = (const float*)d_in[11];
    const int*   esrc = (const int*)d_in[12];
    const int*   edst = (const int*)d_in[13];
    const int*   gids = (const int*)d_in[14];
    float* out = (float*)d_out;

    // workspace carve-up
    char* w = (char*)d_ws;
    float*  h      = (float*)w;  w += (size_t)ND * 4;
    float*  pooled = (float*)w;  w += (size_t)ND * 4;
    float*  z      = (float*)w;  w += (size_t)ND * 4;
    __bf16* Abf    = (__bf16*)w; w += (size_t)ND * 2;
    __bf16* Wbf    = (__bf16*)w; w += (size_t)DD * DD * 2;
    float*  stats  = (float*)w;  w += (size_t)2 * DD * 4;
    float*  gpool  = (float*)w;  w += (size_t)GG * DD * 4;

    const int TB = 256;
    const int gridND   = ND / TB;            // 25000
    const int gridEdge = (EE * DD) / TB;     // 400000
    const int Mtiles   = NN / 16;            // 3125
    const int gemmGrid = (Mtiles * 8) / 4;   // 6250 blocks of 4 waves

    // out = 0
    k_zero<<<(GG * CC + TB - 1) / TB, TB, 0, stream>>>(out, GG * CC);

    // h = x
    hipMemcpyAsync(h, x, (size_t)ND * 4, hipMemcpyDeviceToDevice, stream);

    // readout layer 0 (input features)
    k_zero<<<(GG * DD + TB - 1) / TB, TB, 0, stream>>>(gpool, GG * DD);
    k_graph_pool<<<gridND, TB, 0, stream>>>(h, gids, gpool);
    k_proj<<<(GG * CC) / TB, TB, 0, stream>>>(gpool, Wp + 0, bp + 0, out);

    for (int l = 0; l < LL - 1; ++l) {
        // ---- neighborhood aggregation ----
        k_init_pooled<<<gridND, TB, 0, stream>>>(h, eps, l, pooled);
        k_edge_scatter<<<gridEdge, TB, 0, stream>>>(h, esrc, edst, pooled);

        // ---- MLP linear 1 (bf16 WMMA) ----
        k_cvt_bf16x4<<<(ND / 4) / TB, TB, 0, stream>>>(pooled, Abf);
        k_cvt_w_transpose<<<(DD * DD) / TB, TB, 0, stream>>>(W1 + (size_t)l * DD * DD, Wbf);
        k_gemm_bf16<<<gemmGrid, 128, 0, stream>>>(Abf, Wbf, b1 + l * DD, z, Mtiles);

        // ---- inner BN + ReLU -> bf16 ----
        k_zero<<<1, 256, 0, stream>>>(stats, 2 * DD);
        k_colstats<<<512, 256, 0, stream>>>(z, stats, NN);
        k_bn_finalize<<<1, DD, 0, stream>>>(stats, g1 + l * DD, be1 + l * DD, 1.0f / NN);
        k_bn_relu_bf16<<<gridND, TB, 0, stream>>>(z, stats, Abf);

        // ---- MLP linear 2 (bf16 WMMA) ----
        k_cvt_w_transpose<<<(DD * DD) / TB, TB, 0, stream>>>(W2 + (size_t)l * DD * DD, Wbf);
        k_gemm_bf16<<<gemmGrid, 128, 0, stream>>>(Abf, Wbf, b2 + l * DD, z, Mtiles);

        // ---- outer BN + ReLU -> fp32 h ----
        k_zero<<<1, 256, 0, stream>>>(stats, 2 * DD);
        k_colstats<<<512, 256, 0, stream>>>(z, stats, NN);
        k_bn_finalize<<<1, DD, 0, stream>>>(stats, g2 + l * DD, be2 + l * DD, 1.0f / NN);
        k_bn_relu_f32<<<gridND, TB, 0, stream>>>(z, stats, h);

        // ---- readout for this layer's hidden rep ----
        k_zero<<<(GG * DD + TB - 1) / TB, TB, 0, stream>>>(gpool, GG * DD);
        k_graph_pool<<<gridND, TB, 0, stream>>>(h, gids, gpool);
        k_proj<<<(GG * CC) / TB, TB, 0, stream>>>(gpool,
                                                  Wp + (size_t)(l + 1) * DD * CC,
                                                  bp + (size_t)(l + 1) * CC, out);
    }
}
